// Arch21GraphEncoder_46651934769751
// MI455X (gfx1250) — compile-verified, matrix operand order checked by reading, not run on hardware
//
#include <hip/hip_runtime.h>
#include <hip/hip_bf16.h>

// ---------------------------------------------------------------------------
// MI455X / gfx1250 implementation of the Arch21 graph encoder.
// All dense math runs through a bf16 WMMA GEMM (v_wmma_f32_16x16x32_bf16),
// 128x128x32 workgroup tiles, double-buffered in LDS via the CDNA5 async
// global->LDS path (global_load_async_to_lds_b128 + s_wait_asynccnt), 8 waves
// * (2x4) 16x16 fragments. Epilogues fuse bias/ReLU/residual/bf16/diag.
// ---------------------------------------------------------------------------

typedef __bf16 bf16;
typedef __attribute__((ext_vector_type(16))) __bf16 v16bf;
typedef __attribute__((ext_vector_type(8)))  float  v8f;

#define TM 128
#define TN 128
#define TK 32
#define LST 40   // LDS row stride in bf16 elems (32 + 8 pad, row pitch 80B = 5*16B)

// ---------------- problem constants ----------------
#define HD    256
#define KW    9
#define SB    8192            // B*m subgraphs
#define TOKN  (SB*KW)         // 73728 tokens
#define BG    64
#define NPER  256
#define NN    (BG*NPER)       // 16384 nodes
#define EE    65536           // edges (both directions)
#define MSUB  128

// ---------------- CDNA5 async global->LDS copy (16B per lane) -------------
__device__ __forceinline__ void async_copy_b128(unsigned ldsByteOff, const bf16* gsrc) {
  // GV mode: 64-bit global address in VGPR pair, LDS byte offset in VDST VGPR.
  asm volatile("global_load_async_to_lds_b128 %0, %1, off"
               :: "v"(ldsByteOff), "v"(gsrc) : "memory");
}
__device__ __forceinline__ void wait_async0() {
  asm volatile("s_wait_asynccnt 0" ::: "memory");
}

// ---------------- block-wide sum (256 threads, wave32) ----------------
__device__ __forceinline__ float block_sum_256(float v) {
  __shared__ float red[8];
  __shared__ float tot;
  #pragma unroll
  for (int m = 16; m >= 1; m >>= 1) v += __shfl_xor(v, m, 32);
  int lane = threadIdx.x & 31, w = threadIdx.x >> 5;
  if (lane == 0) red[w] = v;
  __syncthreads();
  if (threadIdx.x == 0) {
    float s = 0.f;
    #pragma unroll
    for (int i = 0; i < 8; i++) s += red[i];
    tot = s;
  }
  __syncthreads();
  float r = tot;
  __syncthreads();
  return r;
}

// ---------------- generic bf16 WMMA GEMM ----------------
// C[b] = A[b] (MxK, row major) * B[b] (KxN row major; NxK if TRANSB)
// epilogue: + bias, +resid, relu, write f32 and/or bf16, optional diag tap.
template <int TRANSB>
__global__ __launch_bounds__(256) void gemm_bf16_wmma(
    const bf16* __restrict__ Ag, const bf16* __restrict__ Bg,
    int M, int N, int K,
    long long sA, long long sB, long long sC,
    const float* __restrict__ bias, int relu,
    const float* __restrict__ resid,
    float* __restrict__ Cf, bf16* __restrict__ Cb,
    float* __restrict__ diagOut, int diagT, int diagIdx)
{
  __shared__ bf16 lsA[2][TM * LST];
  __shared__ bf16 lsB[2][TN * LST];
  const int tid  = threadIdx.x;
  const int lane = tid & 31;
  const int wave = tid >> 5;
  const int wm = wave >> 1, wn = wave & 1;      // 4x2 wave grid
  const int half = lane >> 4, l16 = lane & 15;
  const int batch = blockIdx.z;
  const bf16* A = Ag + (long long)batch * sA;
  const bf16* B = Bg + (long long)batch * sB;
  const int m0 = blockIdx.y * TM, n0 = blockIdx.x * TN;

  v8f acc[2][4];
  #pragma unroll
  for (int i = 0; i < 2; i++)
    #pragma unroll
    for (int j = 0; j < 4; j++)
      #pragma unroll
      for (int r = 0; r < 8; r++) acc[i][j][r] = 0.f;

  // stage tile k0 into LDS buffer `buf`
  auto stage = [&](int buf, int k0) {
    // A tile 128x32 (row major, K contiguous): pure async DMA copy
    #pragma unroll
    for (int c = 0; c < 2; c++) {
      int idx = tid + c * 256;
      int row = idx >> 2, ck = (idx & 3) * 8;
      const bf16* src = A + (long long)(m0 + row) * K + k0 + ck;
      unsigned dst = (unsigned)(size_t)&lsA[buf][row * LST + ck];
      async_copy_b128(dst, src);
    }
    if (TRANSB) {
      // B stored [N][K]: direct async DMA copy into BT[n][k]
      #pragma unroll
      for (int c = 0; c < 2; c++) {
        int idx = tid + c * 256;
        int row = idx >> 2, ck = (idx & 3) * 8;
        const bf16* src = B + (long long)(n0 + row) * K + k0 + ck;
        unsigned dst = (unsigned)(size_t)&lsB[buf][row * LST + ck];
        async_copy_b128(dst, src);
      }
    } else {
      // B stored [K][N]: transpose through VGPRs into BT[n][k]
      #pragma unroll
      for (int c = 0; c < 2; c++) {
        int idx = tid + c * 256;
        int kk = idx >> 4, nn = (idx & 15) * 8;
        const bf16* src = B + (long long)(k0 + kk) * N + n0 + nn;
        uint4 v = *(const uint4*)src;
        bf16 tmp[8]; *(uint4*)tmp = v;
        #pragma unroll
        for (int i = 0; i < 8; i++) lsB[buf][(nn + i) * LST + kk] = tmp[i];
      }
    }
  };

  const int nt = K / TK;
  stage(0, 0);
  for (int i = 0; i < nt; i++) {
    const int cur = i & 1;
    wait_async0();            // my async copies for tile i have landed in LDS
    __syncthreads();          // everyone's landed; prev buffer fully consumed
    if (i + 1 < nt) stage(1 - cur, (i + 1) * TK);   // prefetch overlaps WMMA

    union FragU { v16bf v; uint4 q[2]; };
    FragU aF[2], bF[4];
    #pragma unroll
    for (int fm = 0; fm < 2; fm++) {
      int r = wm * 32 + fm * 16 + l16;
      aF[fm].q[0] = *(const uint4*)&lsA[cur][r * LST + half * 8];
      aF[fm].q[1] = *(const uint4*)&lsA[cur][r * LST + half * 8 + 16];
    }
    #pragma unroll
    for (int fn = 0; fn < 4; fn++) {
      int r = wn * 64 + fn * 16 + l16;
      bF[fn].q[0] = *(const uint4*)&lsB[cur][r * LST + half * 8];
      bF[fn].q[1] = *(const uint4*)&lsB[cur][r * LST + half * 8 + 16];
    }
    #pragma unroll
    for (int fm = 0; fm < 2; fm++)
      #pragma unroll
      for (int fn = 0; fn < 4; fn++)
        acc[fm][fn] = __builtin_amdgcn_wmma_f32_16x16x32_bf16(
            false, aF[fm].v, false, bF[fn].v, (short)0, acc[fm][fn],
            false, false);
  }

  // epilogue
  #pragma unroll
  for (int fm = 0; fm < 2; fm++) {
    #pragma unroll
    for (int fn = 0; fn < 4; fn++) {
      int col = n0 + wn * 64 + fn * 16 + l16;
      #pragma unroll
      for (int r = 0; r < 8; r++) {
        int row = m0 + wm * 32 + fm * 16 + half * 8 + r;
        float v = acc[fm][fn][r];
        if (bias) v += bias[col];
        long long o = (long long)batch * sC + (long long)row * N + col;
        if (resid) v += resid[o];
        if (relu) v = fmaxf(v, 0.f);
        if (Cf) Cf[o] = v;
        if (Cb) Cb[o] = (bf16)v;
        if (diagOut && row == col)
          diagOut[((long long)batch * M + row) * diagT + diagIdx] = v;
      }
    }
  }
}

// ---------------- small utility kernels ----------------
__global__ void zero_f32_k(float* p, long long n) {
  long long i = (long long)blockIdx.x * 256 + threadIdx.x;
  if (i < n) p[i] = 0.f;
}
__global__ void zero_bf16_k(bf16* p, long long n) {
  long long i = (long long)blockIdx.x * 256 + threadIdx.x;
  if (i < n) p[i] = (bf16)0.f;
}
__global__ void cvt_f32_bf16_k(const float* __restrict__ in, bf16* __restrict__ out, long long n) {
  long long i = (long long)blockIdx.x * 256 + threadIdx.x;
  if (i < n) out[i] = (bf16)in[i];
}

// ---------------- RWSE ----------------
__global__ void scatter_edges_k(const int* __restrict__ ei, float* __restrict__ A, int E) {
  int i = blockIdx.x * 256 + threadIdx.x;
  if (i >= E) return;
  int s = ei[i], d = ei[E + i];
  int g = s >> 8, li = s & 255, lj = d & 255;
  atomicAdd(&A[(((long long)(g * 256 + li)) << 8) + lj], 1.f);
}

__global__ __launch_bounds__(256) void normalize_p_k(
    const float* __restrict__ A, bf16* __restrict__ P, bf16* __restrict__ M,
    float* __restrict__ rwse)
{
  int row = blockIdx.x;          // g*256 + li
  int j = threadIdx.x;
  float a = A[(((long long)row) << 8) + j];
  float s = block_sum_256(a);
  float p = a / fmaxf(s, 1.f);
  P[(((long long)row) << 8) + j] = (bf16)p;
  M[(((long long)row) << 8) + j] = (bf16)p;
  if (j == (row & 255)) rwse[(long long)row * 8] = p;   // diag of P = rwse[:,0]
}

// ---------------- gather + embed + RWSE-MLP -> concat bf16 [TOKN][768] ----
__global__ __launch_bounds__(256) void build_hcat_k(
    const float* __restrict__ atom_emb, const float* __restrict__ bond_emb,
    const float* __restrict__ rwse, const float* __restrict__ rwse_W,
    const float* __restrict__ rwse_b,
    const int* __restrict__ x_atoms, const int* __restrict__ edge_attr,
    const int* __restrict__ nodes, const int* __restrict__ esg,
    bf16* __restrict__ hcat, float* __restrict__ validf)
{
  int tok = blockIdx.x; int h = threadIdx.x;
  int s = tok / KW, j = tok - s * KW;
  int raw = nodes[tok];
  int valid = raw >= 0;
  int node = valid ? raw : 0;
  float a = atom_emb[(long long)x_atoms[node] * HD + h];
  float r = rwse_b[h];
  #pragma unroll
  for (int t = 0; t < 8; t++) r += rwse[(long long)node * 8 + t] * rwse_W[t * HD + h];
  r = fmaxf(r, 0.f);
  float bnd = 0.f;
  if (j > 0) {
    int e = esg[s * (KW - 1) + (j - 1)];
    bnd = bond_emb[(long long)(edge_attr[e] - 1) * HD + h];
  }
  long long base = (long long)tok * (3 * HD);
  hcat[base + h]            = (bf16)a;
  hcat[base + HD + h]       = (bf16)r;
  hcat[base + 2 * HD + h]   = (bf16)bnd;
  if (h == 0) validf[tok] = (float)valid;
}

__global__ void mask_h_k(const float* __restrict__ hp, const float* __restrict__ validf,
                         float* __restrict__ h0) {
  int tok = blockIdx.x; int h = threadIdx.x;
  h0[(long long)tok * HD + h] = hp[(long long)tok * HD + h] * validf[tok];
}

__global__ void denom_k(const float* __restrict__ validf, float* __restrict__ denom) {
  int s = blockIdx.x * 256 + threadIdx.x;
  if (s >= SB) return;
  float sum = 0.f;
  #pragma unroll
  for (int j = 0; j < KW; j++) sum += validf[s * KW + j];
  denom[s] = fmaxf(sum, 1.f);
}

// ---------------- GRU gate (adj row computed on the fly) ----------------
__global__ __launch_bounds__(256) void gru_gate_k(
    const float* __restrict__ gh, const int* __restrict__ nodes,
    const float* __restrict__ gW_ih, const float* __restrict__ gb_ih,
    float* __restrict__ hst, bf16* __restrict__ hbf,
    float* __restrict__ peacc, int t)
{
  int s = blockIdx.x; int h = threadIdx.x;
  int nst = max(nodes[s * KW + t], 0);
  float xr[KW];
  #pragma unroll
  for (int c = 0; c < KW; c++) xr[c] = (max(nodes[s * KW + c], 0) == nst) ? 1.f : 0.f;
  float gi_r = gb_ih[h], gi_z = gb_ih[HD + h], gi_n = gb_ih[2 * HD + h];
  #pragma unroll
  for (int c = 0; c < KW; c++) {
    gi_r += xr[c] * gW_ih[h * KW + c];
    gi_z += xr[c] * gW_ih[(HD + h) * KW + c];
    gi_n += xr[c] * gW_ih[(2 * HD + h) * KW + c];
  }
  long long gb = (long long)s * (3 * HD);
  float gh_r = gh[gb + h], gh_z = gh[gb + HD + h], gh_n = gh[gb + 2 * HD + h];
  float hp = hst[(long long)s * HD + h];
  float rr = 1.f / (1.f + __expf(-(gi_r + gh_r)));
  float zz = 1.f / (1.f + __expf(-(gi_z + gh_z)));
  float ng = tanhf(gi_n + rr * gh_n);
  float hn = (1.f - zz) * ng + zz * hp;
  hst[(long long)s * HD + h] = hn;
  hbf[(long long)s * HD + h] = (bf16)hn;
  float vt = (nodes[s * KW + t] >= 0) ? 1.f : 0.f;
  peacc[(long long)s * HD + h] += vt * hn;
}

__global__ void pe_mean_k(const float* __restrict__ peacc, const float* __restrict__ denom,
                          bf16* __restrict__ pmb) {
  int s = blockIdx.x; int h = threadIdx.x;
  pmb[(long long)s * HD + h] = (bf16)(peacc[(long long)s * HD + h] / denom[s]);
}

// ---------------- LayerNorm (one block per token) ----------------
__global__ __launch_bounds__(256) void layernorm_k(
    const float* __restrict__ x, const float* __restrict__ g, const float* __restrict__ b,
    bf16* __restrict__ outb, float* __restrict__ outf)
{
  long long tok = blockIdx.x; int h = threadIdx.x;
  float v = x[tok * HD + h];
  float mu = block_sum_256(v) * (1.f / HD);
  float d = v - mu;
  float var = block_sum_256(d * d) * (1.f / HD);
  float y = d * rsqrtf(var + 1e-5f) * g[h] + b[h];
  if (outb) outb[tok * HD + h] = (bf16)y;
  if (outf) outf[tok * HD + h] = y;
}

// ---------------- attention: one wave per (seq, head, query) ----------------
template <int LMAX>
__global__ __launch_bounds__(256) void attention_k(
    const bf16* __restrict__ Q, const bf16* __restrict__ Kt,
    const bf16* __restrict__ V, bf16* __restrict__ O,
    int numSeq, int L)
{
  int wid = blockIdx.x * 8 + (threadIdx.x >> 5);
  int lane = threadIdx.x & 31;
  int total = numSeq * 4 * L;
  if (wid >= total) return;
  int l = wid % L;
  int head = (wid / L) & 3;
  int seq = wid / (L * 4);
  long long base = (long long)seq * L * HD + head * 64;
  float q0 = (float)Q[base + (long long)l * HD + lane];
  float q1 = (float)Q[base + (long long)l * HD + lane + 32];
  float sc[LMAX];
  float mx = -1e30f;
  for (int m = 0; m < L; m++) {
    float k0 = (float)Kt[base + (long long)m * HD + lane];
    float k1 = (float)Kt[base + (long long)m * HD + lane + 32];
    float d = q0 * k0 + q1 * k1;
    #pragma unroll
    for (int off = 16; off >= 1; off >>= 1) d += __shfl_xor(d, off, 32);
    d *= 0.125f;                 // 1/sqrt(dh=64)
    sc[m] = d;
    mx = fmaxf(mx, d);
  }
  float sum = 0.f;
  for (int m = 0; m < L; m++) { sc[m] = __expf(sc[m] - mx); sum += sc[m]; }
  float inv = 1.f / sum;
  float o0 = 0.f, o1 = 0.f;
  for (int m = 0; m < L; m++) {
    float p = sc[m] * inv;
    o0 += p * (float)V[base + (long long)m * HD + lane];
    o1 += p * (float)V[base + (long long)m * HD + lane + 32];
  }
  O[base + (long long)l * HD + lane]      = (bf16)o0;
  O[base + (long long)l * HD + lane + 32] = (bf16)o1;
}

// ---------------- pooling ----------------
__global__ void combine_k(const float* __restrict__ hln, const float* __restrict__ validf,
                          const float* __restrict__ denom, const float* __restrict__ pef,
                          float* __restrict__ xi0) {
  int s = blockIdx.x; int h = threadIdx.x;
  float acc = 0.f;
  #pragma unroll
  for (int j = 0; j < KW; j++)
    acc += hln[(long long)(s * KW + j) * HD + h] * validf[s * KW + j];
  xi0[(long long)s * HD + h] = acc / denom[s] + pef[(long long)s * HD + h];
}

__global__ void final_mean_k(const float* __restrict__ x, float* __restrict__ out) {
  int b = blockIdx.x; int h = threadIdx.x;
  float acc = 0.f;
  for (int m = 0; m < MSUB; m++)
    acc += x[(long long)(b * MSUB + m) * HD + h];
  out[b * HD + h] = acc * (1.f / MSUB);
}

// ---------------- workspace layout ----------------
static constexpr size_t AL(size_t x) { return (x + 255) & ~size_t(255); }
static constexpr size_t OFF_WA   = 0;                                   // f32 64*65536
static constexpr size_t OFF_PBF  = AL(OFF_WA  + 64ull * 65536 * 4);     // bf16 64*65536
static constexpr size_t OFF_M0   = AL(OFF_PBF + 64ull * 65536 * 2);
static constexpr size_t OFF_M1   = AL(OFF_M0  + 64ull * 65536 * 2);
static constexpr size_t OFF_RWSE = AL(OFF_M1  + 64ull * 65536 * 2);     // f32 NN*8
static constexpr size_t OFF_WBF  = AL(OFF_RWSE + (size_t)NN * 8 * 4);   // bf16 2031616
static constexpr size_t OFF_BIG0 = AL(OFF_WBF + 2031616ull * 2);        // hcat / QKV / ff (151MB)
static constexpr size_t OFF_BIG1 = AL(OFF_BIG0 + 150994944ull);         // h_pre / attnb / x2
static constexpr size_t OFF_BIG2 = AL(OFF_BIG1 + 75497472ull);          // h0 / hln
static constexpr size_t OFF_LNB  = AL(OFF_BIG2 + 75497472ull);          // bf16 TOKN*HD
static constexpr size_t OFF_X1   = AL(OFF_LNB  + 37748736ull);          // f32 TOKN*HD
static constexpr size_t OFF_VAL  = AL(OFF_X1   + 75497472ull);
static constexpr size_t OFF_DEN  = AL(OFF_VAL  + (size_t)TOKN * 4);
static constexpr size_t OFF_GH   = AL(OFF_DEN  + (size_t)SB * 4);       // f32 SB*768
static constexpr size_t OFF_HST  = AL(OFF_GH   + (size_t)SB * 768 * 4);
static constexpr size_t OFF_HBF  = AL(OFF_HST  + (size_t)SB * HD * 4);
static constexpr size_t OFF_PEA  = AL(OFF_HBF  + (size_t)SB * HD * 2);
static constexpr size_t OFF_PMB  = AL(OFF_PEA  + (size_t)SB * HD * 4);
static constexpr size_t OFF_PEF  = AL(OFF_PMB  + (size_t)SB * HD * 2);
static constexpr size_t OFF_XI0  = AL(OFF_PEF  + (size_t)SB * HD * 4);
static constexpr size_t OFF_XI1  = AL(OFF_XI0  + (size_t)SB * HD * 4);
static constexpr size_t OFF_XI2  = AL(OFF_XI1  + (size_t)SB * HD * 4);
static constexpr size_t OFF_LNB2 = AL(OFF_XI2  + (size_t)SB * HD * 4);
static constexpr size_t OFF_Q2   = AL(OFF_LNB2 + (size_t)SB * HD * 2);
static constexpr size_t OFF_K2   = AL(OFF_Q2   + (size_t)SB * HD * 2);
static constexpr size_t OFF_V2   = AL(OFF_K2   + (size_t)SB * HD * 2);
static constexpr size_t OFF_AT2  = AL(OFF_V2   + (size_t)SB * HD * 2);
static constexpr size_t OFF_FF2  = AL(OFF_AT2  + (size_t)SB * HD * 2);  // bf16 SB*1024
static constexpr size_t OFF_XIF  = AL(OFF_FF2  + (size_t)SB * 1024 * 2);

// ---------------- host-side helpers ----------------
static void launch_gemm(hipStream_t st, const bf16* A, const bf16* B,
                        int M, int N, int K, long long sA, long long sB, long long sC,
                        int batches, int transB, const float* bias, int relu,
                        const float* resid, float* Cf, bf16* Cb,
                        float* diag, int diagT, int diagIdx) {
  dim3 g(N / TN, M / TM, batches), b(256);
  if (transB)
    gemm_bf16_wmma<1><<<g, b, 0, st>>>(A, B, M, N, K, sA, sB, sC,
                                       bias, relu, resid, Cf, Cb, diag, diagT, diagIdx);
  else
    gemm_bf16_wmma<0><<<g, b, 0, st>>>(A, B, M, N, K, sA, sB, sC,
                                       bias, relu, resid, Cf, Cb, diag, diagT, diagIdx);
}

struct TfmW {
  const bf16 *Wq, *Wk, *Wv, *Wo, *W1, *W2;
  const float *bq, *bk, *bv, *bo, *b1, *b2;
  const float *ln1g, *ln1b, *ln2g, *ln2b, *lnFg, *lnFb;
};

static void run_tfm(hipStream_t st, const TfmW& w, const float* x0,
                    int T, int numSeq, int L,
                    bf16* lnb, bf16* Qb, bf16* Kb, bf16* Vb, bf16* attnb, bf16* ffb,
                    float* x1, float* x2, float* xOutF) {
  layernorm_k<<<T, 256, 0, st>>>(x0, w.ln1g, w.ln1b, lnb, nullptr);
  launch_gemm(st, lnb, w.Wq, T, HD, HD, 0, 0, 0, 1, 0, w.bq, 0, nullptr, nullptr, Qb, nullptr, 0, 0);
  launch_gemm(st, lnb, w.Wk, T, HD, HD, 0, 0, 0, 1, 0, w.bk, 0, nullptr, nullptr, Kb, nullptr, 0, 0);
  launch_gemm(st, lnb, w.Wv, T, HD, HD, 0, 0, 0, 1, 0, w.bv, 0, nullptr, nullptr, Vb, nullptr, 0, 0);
  int waves = numSeq * 4 * L;
  int blocks = (waves + 7) / 8;
  if (L <= 16)
    attention_k<16><<<blocks, 256, 0, st>>>(Qb, Kb, Vb, attnb, numSeq, L);
  else
    attention_k<128><<<blocks, 256, 0, st>>>(Qb, Kb, Vb, attnb, numSeq, L);
  launch_gemm(st, attnb, w.Wo, T, HD, HD, 0, 0, 0, 1, 0, w.bo, 0, x0, x1, nullptr, nullptr, 0, 0);
  layernorm_k<<<T, 256, 0, st>>>(x1, w.ln2g, w.ln2b, lnb, nullptr);
  launch_gemm(st, lnb, w.W1, T, 4 * HD, HD, 0, 0, 0, 1, 0, w.b1, 1, nullptr, nullptr, ffb, nullptr, 0, 0);
  launch_gemm(st, ffb, w.W2, T, HD, 4 * HD, 0, 0, 0, 1, 0, w.b2, 0, x1, x2, nullptr, nullptr, 0, 0);
  layernorm_k<<<T, 256, 0, st>>>(x2, w.lnFg, w.lnFb, nullptr, xOutF);
}

// d_in layout: JAX pytree flatten of the params dict (keys sorted, capitals
// first), then the remaining inputs in setup_inputs() order:
//  0 atom_emb  1 bond_emb  2 gW_hh  3 gW_ih  4 gb_hh  5 gb_ih  6 in_W  7 in_b
//  8..23 inter {W1,W2,Wk,Wo,Wq,Wv,b1,b2,bk,bo,bq,bv,ln1_b,ln1_g,ln2_b,ln2_g}
//  24 inter_ln_b 25 inter_ln_g  26..41 intra{same}  42 intra_ln_b 43 intra_ln_g
//  44 pe_W 45 pe_b 46 rwse_W 47 rwse_b
//  48 x_atoms 49 edge_attr 50 edge_index 51 ptr 52 batch 53 nodes_sampled 54 edge_src_global
extern "C" void kernel_launch(void* const* d_in, const int* in_sizes, int n_in,
                              void* d_out, int out_size, void* d_ws, size_t ws_size,
                              hipStream_t stream) {
  (void)in_sizes; (void)n_in; (void)out_size; (void)ws_size;
  char* ws = (char*)d_ws;
  hipStream_t st = stream;

  const float* p_atom = (const float*)d_in[0];
  const float* p_bond = (const float*)d_in[1];
  const float* p_gWhh = (const float*)d_in[2];
  const float* p_gWih = (const float*)d_in[3];
  const float* p_gbhh = (const float*)d_in[4];
  const float* p_gbih = (const float*)d_in[5];
  const float* p_inW  = (const float*)d_in[6];
  const float* p_inb  = (const float*)d_in[7];
  const float* p_interlnb = (const float*)d_in[24];
  const float* p_interlng = (const float*)d_in[25];
  const float* p_intralnb = (const float*)d_in[42];
  const float* p_intralng = (const float*)d_in[43];
  const float* p_peW  = (const float*)d_in[44];
  const float* p_peb  = (const float*)d_in[45];
  const float* p_rwW  = (const float*)d_in[46];
  const float* p_rwb  = (const float*)d_in[47];
  const int* p_xat  = (const int*)d_in[48];
  const int* p_eatt = (const int*)d_in[49];
  const int* p_ei   = (const int*)d_in[50];
  const int* p_nod  = (const int*)d_in[53];
  const int* p_esg  = (const int*)d_in[54];

  // bf16 weight arena
  bf16* WBF = (bf16*)(ws + OFF_WBF);
  bf16* w_inW  = WBF + 0;
  bf16* w_gWhh = WBF + 196608;
  bf16* w_peW  = WBF + 393216;
  bf16* iaWq = WBF + 458752,  *iaWk = WBF + 524288,  *iaWv = WBF + 589824;
  bf16* iaWo = WBF + 655360,  *iaW1 = WBF + 720896,  *iaW2 = WBF + 983040;
  bf16* ieWq = WBF + 1245184, *ieWk = WBF + 1310720, *ieWv = WBF + 1376256;
  bf16* ieWo = WBF + 1441792, *ieW1 = WBF + 1507328, *ieW2 = WBF + 1769472;

  auto cvt = [&](const float* src, bf16* dst, long long n) {
    cvt_f32_bf16_k<<<(unsigned)((n + 255) / 256), 256, 0, st>>>(src, dst, n);
  };
  cvt(p_inW,  w_inW,  196608);
  cvt(p_gWhh, w_gWhh, 196608);
  cvt(p_peW,  w_peW,  65536);
  // intra base 26: W1,W2,Wk,Wo,Wq,Wv
  cvt((const float*)d_in[30], iaWq, 65536);
  cvt((const float*)d_in[28], iaWk, 65536);
  cvt((const float*)d_in[31], iaWv, 65536);
  cvt((const float*)d_in[29], iaWo, 65536);
  cvt((const float*)d_in[26], iaW1, 262144);
  cvt((const float*)d_in[27], iaW2, 262144);
  // inter base 8
  cvt((const float*)d_in[12], ieWq, 65536);
  cvt((const float*)d_in[10], ieWk, 65536);
  cvt((const float*)d_in[13], ieWv, 65536);
  cvt((const float*)d_in[11], ieWo, 65536);
  cvt((const float*)d_in[8],  ieW1, 262144);
  cvt((const float*)d_in[9],  ieW2, 262144);

  // ---- RWSE: scatter -> normalize -> 7 batched WMMA power GEMMs ----
  float* wA   = (float*)(ws + OFF_WA);
  bf16*  Pbf  = (bf16*)(ws + OFF_PBF);
  bf16*  Mc   = (bf16*)(ws + OFF_M0);
  bf16*  Mn   = (bf16*)(ws + OFF_M1);
  float* rwse = (float*)(ws + OFF_RWSE);
  zero_f32_k<<<(64 * 65536 + 255) / 256, 256, 0, st>>>(wA, 64ll * 65536);
  scatter_edges_k<<<(EE + 255) / 256, 256, 0, st>>>(p_ei, wA, EE);
  normalize_p_k<<<BG * NPER, 256, 0, st>>>(wA, Pbf, Mc, rwse);
  for (int t = 1; t < 8; t++) {
    launch_gemm(st, Mc, Pbf, NPER, NPER, NPER, 65536, 65536, 65536, BG, 0,
                nullptr, 0, nullptr, nullptr, Mn, rwse, 8, t);
    bf16* tmp = Mc; Mc = Mn; Mn = tmp;
  }

  // ---- gather/embed -> in_W GEMM -> mask ----
  bf16*  hcat   = (bf16*)(ws + OFF_BIG0);
  float* validf = (float*)(ws + OFF_VAL);
  float* denom  = (float*)(ws + OFF_DEN);
  float* h_pre  = (float*)(ws + OFF_BIG1);
  float* h0     = (float*)(ws + OFF_BIG2);
  build_hcat_k<<<TOKN, 256, 0, st>>>(p_atom, p_bond, rwse, p_rwW, p_rwb,
                                     p_xat, p_eatt, p_nod, p_esg, hcat, validf);
  launch_gemm(st, hcat, w_inW, TOKN, HD, 3 * HD, 0, 0, 0, 1, 0,
              p_inb, 0, nullptr, h_pre, nullptr, nullptr, 0, 0);
  mask_h_k<<<TOKN, 256, 0, st>>>(h_pre, validf, h0);
  denom_k<<<(SB + 255) / 256, 256, 0, st>>>(validf, denom);

  // ---- GRU PE: 9x (WMMA GEMM + fused gate) ----
  float* gh    = (float*)(ws + OFF_GH);
  float* hst   = (float*)(ws + OFF_HST);
  bf16*  hbf   = (bf16*)(ws + OFF_HBF);
  float* peacc = (float*)(ws + OFF_PEA);
  bf16*  pmb   = (bf16*)(ws + OFF_PMB);
  float* pef   = (float*)(ws + OFF_PEF);
  zero_f32_k<<<(SB * HD + 255) / 256, 256, 0, st>>>(hst, (long long)SB * HD);
  zero_bf16_k<<<(SB * HD + 255) / 256, 256, 0, st>>>(hbf, (long long)SB * HD);
  zero_f32_k<<<(SB * HD + 255) / 256, 256, 0, st>>>(peacc, (long long)SB * HD);
  for (int t = 0; t < KW; t++) {
    launch_gemm(st, hbf, w_gWhh, SB, 3 * HD, HD, 0, 0, 0, 1, /*transB=*/1,
                p_gbhh, 0, nullptr, gh, nullptr, nullptr, 0, 0);
    gru_gate_k<<<SB, 256, 0, st>>>(gh, p_nod, p_gWih, p_gbih, hst, hbf, peacc, t);
  }
  pe_mean_k<<<SB, 256, 0, st>>>(peacc, denom, pmb);
  launch_gemm(st, pmb, w_peW, SB, HD, HD, 0, 0, 0, 1, 0,
              p_peb, 0, nullptr, pef, nullptr, nullptr, 0, 0);

  // ---- intra transformer (tokens = 73728, L = 9) ----
  bf16*  lnb   = (bf16*)(ws + OFF_LNB);
  bf16*  Qb    = (bf16*)(ws + OFF_BIG0);
  bf16*  Kb    = (bf16*)(ws + OFF_BIG0 + 37748736ull);
  bf16*  Vb    = (bf16*)(ws + OFF_BIG0 + 75497472ull);
  bf16*  attnb = (bf16*)(ws + OFF_BIG1);          // reuses h_pre region
  bf16*  ffb   = (bf16*)(ws + OFF_BIG0);          // reuses QKV region
  float* x1    = (float*)(ws + OFF_X1);
  float* x2    = (float*)(ws + OFF_BIG1);         // attnb dead by then
  float* hln   = (float*)(ws + OFF_BIG2);         // h0 dead by then
  TfmW ia = { iaWq, iaWk, iaWv, iaWo, iaW1, iaW2,
              (const float*)d_in[36], (const float*)d_in[34], (const float*)d_in[37],
              (const float*)d_in[35], (const float*)d_in[32], (const float*)d_in[33],
              (const float*)d_in[39], (const float*)d_in[38],
              (const float*)d_in[41], (const float*)d_in[40],
              p_intralng, p_intralnb };
  run_tfm(st, ia, h0, TOKN, SB, KW, lnb, Qb, Kb, Vb, attnb, ffb, x1, x2, hln);

  // ---- masked pooling + PE ----
  float* xi0 = (float*)(ws + OFF_XI0);
  combine_k<<<SB, 256, 0, st>>>(hln, validf, denom, pef, xi0);

  // ---- inter transformer (tokens = 8192, L = 128) ----
  bf16*  lnb2 = (bf16*)(ws + OFF_LNB2);
  bf16*  Q2 = (bf16*)(ws + OFF_Q2), *K2 = (bf16*)(ws + OFF_K2), *V2 = (bf16*)(ws + OFF_V2);
  bf16*  at2 = (bf16*)(ws + OFF_AT2), *ff2 = (bf16*)(ws + OFF_FF2);
  float* xi1 = (float*)(ws + OFF_XI1);
  float* xi2 = (float*)(ws + OFF_XI2);
  float* xif = (float*)(ws + OFF_XIF);
  TfmW ie = { ieWq, ieWk, ieWv, ieWo, ieW1, ieW2,
              (const float*)d_in[18], (const float*)d_in[16], (const float*)d_in[19],
              (const float*)d_in[17], (const float*)d_in[14], (const float*)d_in[15],
              (const float*)d_in[21], (const float*)d_in[20],
              (const float*)d_in[23], (const float*)d_in[22],
              p_interlng, p_interlnb };
  run_tfm(st, ie, xi0, SB, BG, MSUB, lnb2, Q2, K2, V2, at2, ff2, xi1, xi2, xif);

  // ---- final mean over subgraphs -> [64, 256] f32 ----
  final_mean_k<<<BG, 256, 0, st>>>(xif, (float*)d_out);
}
